// Attention_12532714570579
// MI455X (gfx1250) — matrix operand order repaired and checked by reading, not compiled, not run on hardware
//
#include <hip/hip_runtime.h>

#define DI __device__ __forceinline__

typedef _Float16 f16;
typedef __attribute__((ext_vector_type(16))) _Float16 v16h;
typedef __attribute__((ext_vector_type(8)))  _Float16 v8h;
typedef __attribute__((ext_vector_type(8)))  float    v8f;

// ---------------- problem constants ----------------
constexpr int Q   = 512;
constexpr int Bn  = 16;
constexpr int HID = 768;
constexpr int NH  = 12;
constexpr int HD  = 64;
constexpr int NB  = 63;   // 2*BUCKET-1
constexpr int NBP = 64;   // padded buckets
constexpr int M   = Q * Bn;      // 8192 rows
constexpr int N2  = 2 * HID;     // 1536
constexpr float SCALE = 0.07216878364870323f;  // 1/sqrt(3*HD)
constexpr long long CTX_ELEMS = (long long)M * HID;   // 6291456 floats, then probs

// ---------------- workspace layout (bytes) ----------------
constexpr size_t OFF_HSLN  = 0;                              // f16 [M][HID]  (reused as ctx_ln later)
constexpr size_t SZ_HSLN   = (size_t)M * HID * 2;
constexpr size_t OFF_WQKT  = OFF_HSLN + SZ_HSLN;             // f16 [N2][HID]
constexpr size_t SZ_WQKT   = (size_t)N2 * HID * 2;
constexpr size_t OFF_WVGT  = OFF_WQKT + SZ_WQKT;             // f16 [N2][HID]
constexpr size_t OFF_WOUTT = OFF_WVGT + SZ_WQKT;             // f16 [HID][HID]
constexpr size_t SZ_WOUTT  = (size_t)HID * HID * 2;
constexpr size_t OFF_QM    = OFF_WOUTT + SZ_WOUTT;           // f16 [B][NH][Q][HD]
constexpr size_t SZ_QM     = (size_t)Bn * NH * Q * HD * 2;
constexpr size_t OFF_KM    = OFF_QM + SZ_QM;                 // f16 [B][NH][Q][HD]
constexpr size_t OFF_VT    = OFF_KM + SZ_QM;                 // f16 [B][NH][HD][Q]  (transposed)
constexpr size_t OFF_GATE  = OFF_VT + SZ_QM;                 // f32 [M][HID]
constexpr size_t SZ_GATE   = (size_t)M * HID * 4;
constexpr size_t OFF_QPOS  = OFF_GATE + SZ_GATE;             // f16 [NH][NBP][HD]
constexpr size_t SZ_POS    = (size_t)NH * NBP * HD * 2;
constexpr size_t OFF_KPOS  = OFF_QPOS + SZ_POS;
constexpr size_t OFF_CP    = OFF_KPOS + SZ_POS;              // f32 [B*NH][Q][NBP]
constexpr size_t SZ_CP     = (size_t)Bn * NH * Q * NBP * 4;
constexpr size_t OFF_PCT   = OFF_CP + SZ_CP;                 // f32 [B*NH][Q][NBP]
constexpr size_t OFF_CTX   = OFF_CP;                         // f32 [Q][B][HID]; aliases CP (dead after scores)

// ---------------- WMMA fragment helpers (wave32 layouts per CDNA5 ISA 7.12.2) ----------------
// A 16x32 f16: lane L holds row L&15, K chunks [(L>>4)*8 .. +7] and [+16 .. +23]
DI v16h load_a16(const f16* base, int ld, int lane) {
  const int row = lane & 15, kb = (lane >> 4) * 8;
  const v8h lo = *reinterpret_cast<const v8h*>(base + (size_t)row * ld + kb);
  const v8h hi = *reinterpret_cast<const v8h*>(base + (size_t)row * ld + kb + 16);
  v16h r;
#pragma unroll
  for (int i = 0; i < 8; ++i) { r[i] = lo[i]; r[i + 8] = hi[i]; }
  return r;
}
// B 32x16 f16 (stored as BT[N][K] row-major): lane L holds col L&15, K chunks [(L>>4)*8 .. +7] and [+16 .. +23]
DI v16h load_b16(const f16* baseT, int ld, int lane) {
  const int col = lane & 15, kb = (lane >> 4) * 8;
  const v8h lo = *reinterpret_cast<const v8h*>(baseT + (size_t)col * ld + kb);
  const v8h hi = *reinterpret_cast<const v8h*>(baseT + (size_t)col * ld + kb + 16);
  v16h r;
#pragma unroll
  for (int i = 0; i < 8; ++i) { r[i] = lo[i]; r[i + 8] = hi[i]; }
  return r;
}
// A fragment from an fp32 matrix (probs), converted on the fly
DI v16h load_a32(const float* base, int ld, int lane) {
  const int row = lane & 15, kb = (lane >> 4) * 8;
  const float* p = base + (size_t)row * ld + kb;
  v16h r;
#pragma unroll
  for (int i = 0; i < 8; ++i) { r[i] = (f16)p[i]; r[i + 8] = (f16)p[i + 16]; }
  return r;
}
DI v8f wmma16(v16h a, v16h b, v8f c) {
  return __builtin_amdgcn_wmma_f32_16x16x32_f16(false, a, false, b, (short)0, c, false, false);
}
// 32(M) x 64(N) register tile: 2x4 wmma tiles, K stepped by 32.
// PF: emit global_prefetch hints (worth it only for deep-K loops).
template <bool PF>
DI void mm_2x4_f16(const f16* A, int lda, const f16* BT, int ldb, int K, int lane, v8f acc[2][4]) {
  for (int k0 = 0; k0 < K; k0 += 32) {
    if (PF) {
      __builtin_prefetch((const void*)(A + (size_t)(lane & 15) * lda + k0 + 128), 0, 1);
      __builtin_prefetch((const void*)(BT + (size_t)(lane & 15) * ldb + k0 + 128), 0, 1);
    }
    v16h a[2];
#pragma unroll
    for (int mi = 0; mi < 2; ++mi) a[mi] = load_a16(A + (size_t)mi * 16 * lda + k0, lda, lane);
    v16h bf[4];
#pragma unroll
    for (int ni = 0; ni < 4; ++ni) bf[ni] = load_b16(BT + (size_t)ni * 16 * ldb + k0, ldb, lane);
#pragma unroll
    for (int mi = 0; mi < 2; ++mi)
#pragma unroll
      for (int ni = 0; ni < 4; ++ni)
        acc[mi][ni] = wmma16(a[mi], bf[ni], acc[mi][ni]);
  }
}

// ---------------- K1/K9: row layernorm (768) fp32 -> f16 ----------------
__global__ void k_ln_cast(const float* __restrict__ src, f16* __restrict__ dst) {
  __shared__ float s_sum[256], s_sq[256];
  const int row = blockIdx.x;
  const float* x = src + (size_t)row * HID;
  const float a0 = x[threadIdx.x], a1 = x[threadIdx.x + 256], a2 = x[threadIdx.x + 512];
  s_sum[threadIdx.x] = a0 + a1 + a2;
  s_sq[threadIdx.x]  = a0 * a0 + a1 * a1 + a2 * a2;
  __syncthreads();
  for (int st = 128; st > 0; st >>= 1) {
    if (threadIdx.x < st) {
      s_sum[threadIdx.x] += s_sum[threadIdx.x + st];
      s_sq[threadIdx.x]  += s_sq[threadIdx.x + st];
    }
    __syncthreads();
  }
  const float mu   = s_sum[0] * (1.0f / HID);
  const float var  = s_sq[0] * (1.0f / HID) - mu * mu;
  const float rstd = rsqrtf(var + 1e-7f);
  f16* d = dst + (size_t)row * HID;
  d[threadIdx.x]       = (f16)((a0 - mu) * rstd);
  d[threadIdx.x + 256] = (f16)((a1 - mu) * rstd);
  d[threadIdx.x + 512] = (f16)((a2 - mu) * rstd);
}

// ---------------- K2: transpose + cast weights: fp32 [rows][cols] -> f16 [cols][rows] ----------------
__global__ void k_castT(const float* __restrict__ src, f16* __restrict__ dst, int rows, int cols) {
  const long long i = (long long)blockIdx.x * blockDim.x + threadIdx.x;
  if (i >= (long long)rows * cols) return;
  const int r = (int)(i / cols), c = (int)(i % cols);
  dst[(size_t)c * rows + r] = (f16)src[i];
}

// ---------------- K3: pos = rel_emb @ w_qk + b_qk, scattered to q_pos/k_pos head layout ----------------
__global__ void k_pos(const float* __restrict__ re, const float* __restrict__ w_qk,
                      const float* __restrict__ b_qk, f16* __restrict__ qpos, f16* __restrict__ kpos) {
  const int gid = blockIdx.x * blockDim.x + threadIdx.x;
  if (gid >= NBP * N2) return;
  const int kb = gid / N2, n = gid % N2;
  float acc = 0.0f;
  if (kb < NB) {
    acc = b_qk[n];
    const float* r = re + (size_t)kb * HID;
    for (int k = 0; k < HID; ++k) acc += r[k] * w_qk[(size_t)k * N2 + n];
  }
  const f16 v = (f16)acc;
  if (n < HID) {
    qpos[((size_t)(n >> 6) * NBP + kb) * HD + (n & 63)] = v;
  } else {
    const int n2 = n - HID;
    kpos[((size_t)(n2 >> 6) * NBP + kb) * HD + (n2 & 63)] = v;
  }
}

// ---------------- K4: big projection GEMM (modes: 0=qk scatter, 1=vg scatter+gelu, 2=out) ----------------
__global__ void __launch_bounds__(256, 1)
k_gemm_proj(const f16* __restrict__ A, const f16* __restrict__ BT,
            const float* __restrict__ bias, int Ncols, int mode,
            f16* __restrict__ o0, f16* __restrict__ o1, float* __restrict__ o2) {
  const int lane = threadIdx.x & 31, wid = threadIdx.x >> 5;
  const int wm = wid & 3, wn = wid >> 2;
  const int m0 = blockIdx.y * 128 + wm * 32;
  const int n0 = blockIdx.x * 128 + wn * 64;
  v8f acc[2][4] = {};
  mm_2x4_f16<true>(A + (size_t)m0 * HID, HID, BT + (size_t)n0 * HID, HID, HID, lane, acc);
#pragma unroll
  for (int mi = 0; mi < 2; ++mi)
#pragma unroll
    for (int ni = 0; ni < 4; ++ni)
#pragma unroll
      for (int j = 0; j < 8; ++j) {
        const int gm = m0 + mi * 16 + j + ((lane >> 4) << 3);
        const int gn = n0 + ni * 16 + (lane & 15);
        const float v = acc[mi][ni][j] + bias[gn];
        const int qp = gm >> 4, bb = gm & 15;   // row m = q*16 + b
        if (mode == 0) {
          if (gn < HID) {
            const int h = gn >> 6, d = gn & 63;
            o0[(((size_t)bb * NH + h) * Q + qp) * HD + d] = (f16)v;            // q[b][h][q][d]
          } else {
            const int g = gn - HID, h = g >> 6, d = g & 63;
            o1[(((size_t)bb * NH + h) * Q + qp) * HD + d] = (f16)v;            // k[b][h][k][d]
          }
        } else if (mode == 1) {
          if (gn < HID) {
            const int h = gn >> 6, d = gn & 63;
            o0[(((size_t)bb * NH + h) * HD + d) * Q + qp] = (f16)v;            // vT[b][h][d][k]
          } else {
            const int g = gn - HID;
            const float gl = 0.5f * v * (1.0f + erff(v * 0.70710678118f));     // exact gelu
            o2[(size_t)gm * HID + g] = gl;                                     // gate[q*16+b][hid]
          }
        } else {
          o2[(size_t)gm * Ncols + gn] = v;                                     // final output
        }
      }
}

// ---------------- K5: cp = SCALE*q@kpos^T ; pcT = SCALE*k@qpos^T (per bh, [Q][64]) ----------------
__global__ void __launch_bounds__(256, 1)
k_pos_scores(const f16* __restrict__ qm, const f16* __restrict__ km,
             const f16* __restrict__ qpos, const f16* __restrict__ kpos,
             float* __restrict__ cp, float* __restrict__ pct) {
  const int lane = threadIdx.x & 31, wid = threadIdx.x >> 5;
  const int bh = blockIdx.x >> 1, tile = blockIdx.x & 1;
  const int h = bh % NH;
  const int sel = blockIdx.y;     // 0: cp (A=q, B=kpos)   1: pcT (A=k, B=qpos)
  const int m0 = tile * 256 + wid * 32;
  const f16* A  = (sel ? km : qm) + ((size_t)bh * Q + m0) * HD;
  const f16* BT = (sel ? qpos : kpos) + (size_t)h * NBP * HD;
  v8f acc[2][4] = {};
  mm_2x4_f16<false>(A, HD, BT, HD, HD, lane, acc);
  float* dst = (sel ? pct : cp) + ((size_t)bh * Q + m0) * NBP;
#pragma unroll
  for (int mi = 0; mi < 2; ++mi)
#pragma unroll
    for (int ni = 0; ni < 4; ++ni)
#pragma unroll
      for (int j = 0; j < 8; ++j) {
        const int r = mi * 16 + j + ((lane >> 4) << 3);
        const int c = ni * 16 + (lane & 15);
        dst[(size_t)r * NBP + c] = acc[mi][ni][j] * SCALE;
      }
}

// ---------------- K6: scores = SCALE*q@k^T + gathered cp/pcT, masked ----------------
__global__ void __launch_bounds__(256, 1)
k_scores(const f16* __restrict__ qm, const f16* __restrict__ km,
         const float* __restrict__ cp, const float* __restrict__ pct,
         const int* __restrict__ pidx, const unsigned char* __restrict__ amask,
         float* __restrict__ probs) {
  const int lane = threadIdx.x & 31, wid = threadIdx.x >> 5;
  const int wm = wid & 3, wn = wid >> 2;
  const int bh = blockIdx.z, b = bh / NH;
  const int q0 = blockIdx.y * 128 + wm * 32;
  const int k0 = blockIdx.x * 128 + wn * 64;
  const f16* A  = qm + ((size_t)bh * Q + q0) * HD;
  const f16* BT = km + ((size_t)bh * Q + k0) * HD;   // B col = key index, contiguous over d
  v8f acc[2][4] = {};
  mm_2x4_f16<false>(A, HD, BT, HD, HD, lane, acc);
  const float* cpb = cp  + (size_t)bh * Q * NBP;
  const float* pcb = pct + (size_t)bh * Q * NBP;
  float* out = probs + (size_t)bh * Q * Q;
  const unsigned char* mk = amask + (size_t)b * Q * Q;
#pragma unroll
  for (int mi = 0; mi < 2; ++mi)
#pragma unroll
    for (int ni = 0; ni < 4; ++ni)
#pragma unroll
      for (int j = 0; j < 8; ++j) {
        const int gq = q0 + mi * 16 + j + ((lane >> 4) << 3);
        const int gk = k0 + ni * 16 + (lane & 15);
        const int id = pidx[(size_t)gq * Q + gk];
        float s = acc[mi][ni][j] * SCALE
                + cpb[(size_t)gq * NBP + id]
                + pcb[(size_t)gk * NBP + id];
        if (mk[(size_t)gq * Q + gk]) s = -__builtin_inff();
        out[(size_t)gq * Q + gk] = s;
      }
}

// ---------------- K7: in-place softmax, one wave per 512-row ----------------
__global__ void k_softmax(float* __restrict__ probs) {
  const int lane = threadIdx.x & 31, wid = threadIdx.x >> 5;
  const long long row = (long long)blockIdx.x * 8 + wid;
  float* x = probs + row * Q;
  float vals[16];
  float m = -__builtin_inff();
#pragma unroll
  for (int i = 0; i < 16; ++i) { vals[i] = x[lane + i * 32]; m = fmaxf(m, vals[i]); }
#pragma unroll
  for (int off = 16; off > 0; off >>= 1) m = fmaxf(m, __shfl_xor(m, off, 32));
  float s = 0.0f;
#pragma unroll
  for (int i = 0; i < 16; ++i) { const float e = expf(vals[i] - m); vals[i] = e; s += e; }
#pragma unroll
  for (int off = 16; off > 0; off >>= 1) s += __shfl_xor(s, off, 32);
  const bool ok = (m > -__builtin_inff());
  const float inv = ok ? 1.0f / s : 0.0f;
#pragma unroll
  for (int i = 0; i < 16; ++i) x[lane + i * 32] = ok ? vals[i] * inv : 0.0f;
}

// ---------------- K8: ctx = (probs @ v) * gate, scattered to [Q][B][HID] ----------------
__global__ void __launch_bounds__(256, 1)
k_ctx(const float* __restrict__ probs, const f16* __restrict__ vT,
      const float* __restrict__ gate, float* __restrict__ ctx) {
  const int lane = threadIdx.x & 31, wid = threadIdx.x >> 5;
  const int bh = blockIdx.y, b = bh / NH, h = bh % NH;
  const int q0 = blockIdx.x * 256 + wid * 32;
  const float* A = probs + ((size_t)bh * Q + q0) * Q;
  const f16* BT  = vT + (size_t)bh * HD * Q;          // vT[d][k]: B col = d, contiguous over k
  v8f acc[2][4] = {};
  for (int k0 = 0; k0 < Q; k0 += 32) {
    __builtin_prefetch((const void*)(A + (size_t)(lane & 15) * Q + k0 + 128), 0, 1);
    v16h a[2];
#pragma unroll
    for (int mi = 0; mi < 2; ++mi) a[mi] = load_a32(A + (size_t)mi * 16 * Q + k0, Q, lane);
    v16h bf[4];
#pragma unroll
    for (int ni = 0; ni < 4; ++ni) bf[ni] = load_b16(BT + (size_t)ni * 16 * Q + k0, Q, lane);
#pragma unroll
    for (int mi = 0; mi < 2; ++mi)
#pragma unroll
      for (int ni = 0; ni < 4; ++ni)
        acc[mi][ni] = wmma16(a[mi], bf[ni], acc[mi][ni]);
  }
#pragma unroll
  for (int mi = 0; mi < 2; ++mi)
#pragma unroll
    for (int ni = 0; ni < 4; ++ni)
#pragma unroll
      for (int j = 0; j < 8; ++j) {
        const int gq = q0 + mi * 16 + j + ((lane >> 4) << 3);
        const int d  = ni * 16 + (lane & 15);
        const size_t idx = ((size_t)gq * Bn + b) * HID + h * HD + d;
        ctx[idx] = acc[mi][ni][j] * gate[idx];
      }
}

// ---------------- launcher ----------------
extern "C" void kernel_launch(void* const* d_in, const int* in_sizes, int n_in,
                              void* d_out, int out_size, void* d_ws, size_t ws_size,
                              hipStream_t stream) {
  const float*         hs    = (const float*)d_in[0];
  const unsigned char* amask = (const unsigned char*)d_in[1];   // bool mask
  const float*         re    = (const float*)d_in[2];
  const int*           pidx  = (const int*)d_in[3];
  const float*         w_qk  = (const float*)d_in[4];
  const float*         b_qk  = (const float*)d_in[5];
  const float*         w_vg  = (const float*)d_in[6];
  const float*         b_vg  = (const float*)d_in[7];
  const float*         w_out = (const float*)d_in[8];
  const float*         b_out = (const float*)d_in[9];

  char* ws = (char*)d_ws;
  f16*   hsln  = (f16*)(ws + OFF_HSLN);
  f16*   wqkT  = (f16*)(ws + OFF_WQKT);
  f16*   wvgT  = (f16*)(ws + OFF_WVGT);
  f16*   woutT = (f16*)(ws + OFF_WOUTT);
  f16*   qm    = (f16*)(ws + OFF_QM);
  f16*   km    = (f16*)(ws + OFF_KM);
  f16*   vT    = (f16*)(ws + OFF_VT);
  float* gate  = (float*)(ws + OFF_GATE);
  f16*   qpos  = (f16*)(ws + OFF_QPOS);
  f16*   kpos  = (f16*)(ws + OFF_KPOS);
  float* cp    = (float*)(ws + OFF_CP);
  float* pct   = (float*)(ws + OFF_PCT);
  float* ctx   = (float*)(ws + OFF_CTX);   // aliases cp (dead after k_scores)

  float* outp  = (float*)d_out;
  float* probs = outp + CTX_ELEMS;

  k_ln_cast<<<M, 256, 0, stream>>>(hs, hsln);
  k_castT<<<(HID * N2 + 255) / 256, 256, 0, stream>>>(w_qk, wqkT, HID, N2);
  k_castT<<<(HID * N2 + 255) / 256, 256, 0, stream>>>(w_vg, wvgT, HID, N2);
  k_castT<<<(HID * HID + 255) / 256, 256, 0, stream>>>(w_out, woutT, HID, HID);
  k_pos<<<(NBP * N2 + 255) / 256, 256, 0, stream>>>(re, w_qk, b_qk, qpos, kpos);

  k_gemm_proj<<<dim3(N2 / 128, M / 128), 256, 0, stream>>>(hsln, wqkT, b_qk, N2, 0, qm, km, nullptr);
  k_gemm_proj<<<dim3(N2 / 128, M / 128), 256, 0, stream>>>(hsln, wvgT, b_vg, N2, 1, vT, nullptr, gate);

  k_pos_scores<<<dim3(Bn * NH * 2, 2), 256, 0, stream>>>(qm, km, qpos, kpos, cp, pct);
  k_scores<<<dim3(4, 4, Bn * NH), 256, 0, stream>>>(qm, km, cp, pct, pidx, amask, probs);
  k_softmax<<<Bn * NH * Q / 8, 256, 0, stream>>>(probs);
  k_ctx<<<dim3(2, Bn * NH), 256, 0, stream>>>(probs, vT, gate, ctx);

  k_ln_cast<<<M, 256, 0, stream>>>(ctx, hsln);   // reuse hsln buffer as ctx_ln
  k_gemm_proj<<<dim3(HID / 128, M / 128), 256, 0, stream>>>(hsln, woutT, b_out, HID, 2,
                                                            nullptr, nullptr, outp);
}
